// EquivalentTransformer_33277406609693
// MI455X (gfx1250) — matrix operand order repaired and checked by reading, not compiled, not run on hardware
//
#include <hip/hip_runtime.h>

#define N_NODES 100000
#define N_EDGES 1600000
#define EK_WAVES 4
#define NK_WAVES 4
#define OK_WAVES 4
#define NODE_TILES (N_NODES / 16)   // 6250, exact

typedef __attribute__((ext_vector_type(16))) __bf16 v16bf;
typedef __attribute__((ext_vector_type(8)))  __bf16 v8bf;
typedef __attribute__((ext_vector_type(8)))  float  v8f;

__device__ __forceinline__ __bf16 f2bf(float f) {
  unsigned u = __builtin_bit_cast(unsigned, f);
  unsigned r = u + 0x7FFFu + ((u >> 16) & 1u);          // RNE
  unsigned short h = (unsigned short)(r >> 16);
  return __builtin_bit_cast(__bf16, h);
}
__device__ __forceinline__ float bf2f(__bf16 b) {
  unsigned short h = __builtin_bit_cast(unsigned short, b);
  unsigned u = ((unsigned)h) << 16;
  return __builtin_bit_cast(float, u);
}
__device__ __forceinline__ float silu_f(float x) { return x / (1.f + __expf(-x)); }
__device__ __forceinline__ float gelu_f(float x) { return 0.5f * x * (1.f + erff(x * 0.70710678118654752f)); }

__device__ __forceinline__ v8f wmma_bf16(v16bf a, v16bf b, v8f c) {
  return __builtin_amdgcn_wmma_f32_16x16x32_bf16(false, a, false, b, (short)0, c, false, false);
}

// A fragment: rows 0..15 of a row-major bf16 LDS tile, K window [kBase, kBase+32).
// lanes 0-15: row=lane, K = kBase+{0..7,16..23}; lanes 16-31: row=lane-15? no: row=lane&15, K +8.
__device__ __forceinline__ v16bf load_a_frag(const __bf16* tile, int stride, int kBase) {
  int lane = threadIdx.x & 31;
  const __bf16* p = tile + (lane & 15) * stride + kBase + ((lane >> 4) << 3);
  v8bf lo = *(const v8bf*)p;
  v8bf hi = *(const v8bf*)(p + 16);
  return __builtin_shufflevector(lo, hi, 0,1,2,3,4,5,6,7,8,9,10,11,12,13,14,15);
}
// B fragment from an N-major (transposed) weight tile wT[n][k]:
// lanes 0-15 hold K=kBase+{0..7,16..23}, lanes 16-31 hold K=kBase+{8..15,24..31}, col N = nBase + (lane&15).
__device__ __forceinline__ v16bf load_b_frag(const __bf16* wT, int stride, int nBase, int kBase) {
  int lane = threadIdx.x & 31;
  const __bf16* p = wT + (nBase + (lane & 15)) * stride + kBase + ((lane >> 4) << 3);
  v8bf lo = *(const v8bf*)p;
  v8bf hi = *(const v8bf*)(p + 16);
  return __builtin_shufflevector(lo, hi, 0,1,2,3,4,5,6,7,8,9,10,11,12,13,14,15);
}

__global__ void zero_kernel(float* __restrict__ p, int n) {
  int i = blockIdx.x * blockDim.x + threadIdx.x;
  if (i < n) p[i] = 0.f;
}

// ---------------- edge kernel: one wave = 16 edges ----------------
__global__ __launch_bounds__(128) void egnn_edge_kernel(
    const float* __restrict__ h, const float* __restrict__ x,
    const int* __restrict__ src, const int* __restrict__ dst,
    const float* __restrict__ ew1, const float* __restrict__ eb1,
    const float* __restrict__ ew2, const float* __restrict__ eb2,
    const float* __restrict__ cw1, const float* __restrict__ cb1,
    const float* __restrict__ cw2,
    float* __restrict__ hn, float* __restrict__ xsum, float* __restrict__ deg) {
  __shared__ __align__(16) __bf16 w1T[64 * 136];   // edge_w1 rows 0..127, transposed
  __shared__ __align__(16) __bf16 w2T[64 * 72];
  __shared__ __align__(16) __bf16 c1T[64 * 72];
  __shared__ float w128s[64], eb1s[64], eb2s[64], cb1s[64], cw2s[64];
  __shared__ __align__(16) __bf16 fT[EK_WAVES][16 * 136];  // F tile, later msg_h (cols 0..63)
  __shared__ __align__(16) __bf16 yT[EK_WAVES][16 * 72];   // Y1 tile, later Y3
  __shared__ int sidx[EK_WAVES][16], didx[EK_WAVES][16];
  __shared__ float xdv[EK_WAVES][16][3];
  __shared__ float rads[EK_WAVES][16];

  const int tid = threadIdx.x;
  for (int i = tid; i < 129 * 64; i += 128) {
    int k = i >> 6, n = i & 63;
    float v = ew1[i];
    if (k < 128) w1T[n * 136 + k] = f2bf(v); else w128s[n] = v;   // radial row kept f32
  }
  for (int i = tid; i < 64 * 64; i += 128) {
    int k = i >> 6, n = i & 63;
    w2T[n * 72 + k] = f2bf(ew2[i]);
    c1T[n * 72 + k] = f2bf(cw1[i]);
  }
  if (tid < 64) { eb1s[tid] = eb1[tid]; eb2s[tid] = eb2[tid]; cb1s[tid] = cb1[tid]; cw2s[tid] = cw2[tid]; }
  __syncthreads();

  const int wave = tid >> 5, lane = tid & 31;
  const int laneRow = lane & 15, laneHi = lane >> 4;
  const int tile = blockIdx.x * EK_WAVES + wave;
  const int eBase = tile * 16;
  __bf16* f = fT[wave];
  __bf16* y = yT[wave];
  int* si = sidx[wave];
  int* di = didx[wave];

  if (lane < 16) {
    int e = eBase + lane;
    int s = src[e], d = dst[e];
    si[lane] = s; di[lane] = d;
    float dx0 = x[s * 3 + 0] - x[d * 3 + 0];
    float dx1 = x[s * 3 + 1] - x[d * 3 + 1];
    float dx2 = x[s * 3 + 2] - x[d * 3 + 2];
    float r = dx0 * dx0 + dx1 * dx1 + dx2 * dx2;
    float inv = 1.f / (sqrtf(r) + 1e-30f);
    xdv[wave][lane][0] = dx0 * inv;
    xdv[wave][lane][1] = dx1 * inv;
    xdv[wave][lane][2] = dx2 * inv;
    rads[wave][lane] = r;
  }
  __builtin_amdgcn_wave_barrier();

  for (int j = 0; j < 16; ++j) {
    int s = si[j], d = di[j];
    f[j * 136 + lane]       = f2bf(h[s * 64 + lane]);
    f[j * 136 + 32 + lane]  = f2bf(h[s * 64 + 32 + lane]);
    f[j * 136 + 64 + lane]  = f2bf(h[d * 64 + lane]);
    f[j * 136 + 96 + lane]  = f2bf(h[d * 64 + 32 + lane]);
  }
  __builtin_amdgcn_wave_barrier();

  // GEMM1: Y1 = silu(F @ W1[0:128] + radial (x) W1[128] + b1)
  {
    v16bf a[4];
#pragma unroll
    for (int kt = 0; kt < 4; ++kt) a[kt] = load_a_frag(f, 136, kt * 32);
#pragma unroll
    for (int nt = 0; nt < 4; ++nt) {
      v8f acc = {};
#pragma unroll
      for (int kt = 0; kt < 4; ++kt) acc = wmma_bf16(a[kt], load_b_frag(w1T, 136, nt * 16, kt * 32), acc);
      int col = nt * 16 + laneRow;
      float wr = w128s[col], b = eb1s[col];
#pragma unroll
      for (int i = 0; i < 8; ++i) {
        int row = i + laneHi * 8;
        float v = acc[i] + rads[wave][row] * wr + b;
        y[row * 72 + col] = f2bf(silu_f(v));
      }
    }
  }
  __builtin_amdgcn_wave_barrier();

  // GEMM2: msg_h = silu(Y1 @ W2 + b2)  -> stored into f (cols 0..63)
  {
    v16bf a[2];
#pragma unroll
    for (int kt = 0; kt < 2; ++kt) a[kt] = load_a_frag(y, 72, kt * 32);
#pragma unroll
    for (int nt = 0; nt < 4; ++nt) {
      v8f acc = {};
#pragma unroll
      for (int kt = 0; kt < 2; ++kt) acc = wmma_bf16(a[kt], load_b_frag(w2T, 72, nt * 16, kt * 32), acc);
      int col = nt * 16 + laneRow;
      float b = eb2s[col];
#pragma unroll
      for (int i = 0; i < 8; ++i) {
        int row = i + laneHi * 8;
        f[row * 136 + col] = f2bf(silu_f(acc[i] + b));
      }
    }
  }
  __builtin_amdgcn_wave_barrier();

  // GEMM3: Y3 = silu(msg_h @ CW1 + cb1) -> y
  {
    v16bf a[2];
#pragma unroll
    for (int kt = 0; kt < 2; ++kt) a[kt] = load_a_frag(f, 136, kt * 32);
#pragma unroll
    for (int nt = 0; nt < 4; ++nt) {
      v8f acc = {};
#pragma unroll
      for (int kt = 0; kt < 2; ++kt) acc = wmma_bf16(a[kt], load_b_frag(c1T, 72, nt * 16, kt * 32), acc);
      int col = nt * 16 + laneRow;
      float b = cb1s[col];
#pragma unroll
      for (int i = 0; i < 8; ++i) {
        int row = i + laneHi * 8;
        y[row * 72 + col] = f2bf(silu_f(acc[i] + b));
      }
    }
  }
  __builtin_amdgcn_wave_barrier();

  // coord scalar + x scatter (lanes 0..15, one edge each)
  if (lane < 16) {
    float s = 0.f;
    for (int k = 0; k < 64; ++k) s += bf2f(y[lane * 72 + k]) * cw2s[k];
    int d = di[lane];
    atomicAdd(&xsum[d * 3 + 0], s * xdv[wave][lane][0]);
    atomicAdd(&xsum[d * 3 + 1], s * xdv[wave][lane][1]);
    atomicAdd(&xsum[d * 3 + 2], s * xdv[wave][lane][2]);
    atomicAdd(&deg[d], 1.f);
  }
  // msg_h scatter: 16x64 values over 32 lanes
#pragma unroll
  for (int i = 0; i < 32; ++i) {
    int flat = i * 32 + lane;
    int row = flat >> 6, colc = flat & 63;
    atomicAdd(&hn[di[row] * 64 + colc], bf2f(f[row * 136 + colc]));
  }
}

// ---------------- node kernel: one wave = 16 nodes ----------------
__global__ __launch_bounds__(128) void egnn_node_kernel(
    const float* __restrict__ h_in, const float* __restrict__ x_in,
    const float* __restrict__ hn, const float* __restrict__ xsum, const float* __restrict__ deg,
    const float* __restrict__ nw1, const float* __restrict__ nb1,
    const float* __restrict__ nw2, const float* __restrict__ nb2,
    const float* __restrict__ lng, const float* __restrict__ lnb,
    float* __restrict__ h_out, float* __restrict__ x_out) {
  __shared__ __align__(16) __bf16 n1T[64 * 136];
  __shared__ __align__(16) __bf16 n2T[64 * 72];
  __shared__ float nb1s[64], nb2s[64], lngs[64], lnbs[64];
  __shared__ __align__(16) __bf16 aT[NK_WAVES][16 * 136];  // A tile; reused as f32 [16][68] out tile
  __shared__ __align__(16) __bf16 yT[NK_WAVES][16 * 72];
  __shared__ float mus[NK_WAVES][16], rss[NK_WAVES][16];

  const int tid = threadIdx.x;
  for (int i = tid; i < 128 * 64; i += 128) {
    int k = i >> 6, n = i & 63;
    n1T[n * 136 + k] = f2bf(nw1[i]);
  }
  for (int i = tid; i < 64 * 64; i += 128) {
    int k = i >> 6, n = i & 63;
    n2T[n * 72 + k] = f2bf(nw2[i]);
  }
  if (tid < 64) { nb1s[tid] = nb1[tid]; nb2s[tid] = nb2[tid]; lngs[tid] = lng[tid]; lnbs[tid] = lnb[tid]; }
  __syncthreads();

  const int wave = tid >> 5, lane = tid & 31;
  const int laneRow = lane & 15, laneHi = lane >> 4;
  const int tile = blockIdx.x * NK_WAVES + wave;
  if (tile >= NODE_TILES) return;
  const int rBase = tile * 16;
  __bf16* a_tile = aT[wave];
  __bf16* y = yT[wave];
  float* o = (float*)a_tile;  // overlay: 16*68 f32 == 16*136 bf16 bytes

  if (lane < 16) {
    int node = rBase + lane;
    float dg = fmaxf(deg[node], 1.f);
#pragma unroll
    for (int c = 0; c < 3; ++c)
      x_out[node * 3 + c] = x_in[node * 3 + c] + xsum[node * 3 + c] / dg;
  }

  for (int j = 0; j < 16; ++j) {
    int node = rBase + j;
    a_tile[j * 136 + lane]      = f2bf(h_in[node * 64 + lane]);
    a_tile[j * 136 + 32 + lane] = f2bf(h_in[node * 64 + 32 + lane]);
    a_tile[j * 136 + 64 + lane] = f2bf(hn[node * 64 + lane]);
    a_tile[j * 136 + 96 + lane] = f2bf(hn[node * 64 + 32 + lane]);
  }
  __builtin_amdgcn_wave_barrier();

  // GEMM1: silu([h|hn] @ NW1 + b1)
  {
    v16bf a[4];
#pragma unroll
    for (int kt = 0; kt < 4; ++kt) a[kt] = load_a_frag(a_tile, 136, kt * 32);
#pragma unroll
    for (int nt = 0; nt < 4; ++nt) {
      v8f acc = {};
#pragma unroll
      for (int kt = 0; kt < 4; ++kt) acc = wmma_bf16(a[kt], load_b_frag(n1T, 136, nt * 16, kt * 32), acc);
      int col = nt * 16 + laneRow;
      float b = nb1s[col];
#pragma unroll
      for (int i = 0; i < 8; ++i) {
        int row = i + laneHi * 8;
        y[row * 72 + col] = f2bf(silu_f(acc[i] + b));
      }
    }
  }
  __builtin_amdgcn_wave_barrier();

  // GEMM2 + gelu -> f32 tile (overlaying A tile)
  {
    v16bf a[2];
#pragma unroll
    for (int kt = 0; kt < 2; ++kt) a[kt] = load_a_frag(y, 72, kt * 32);
    __builtin_amdgcn_wave_barrier();
#pragma unroll
    for (int nt = 0; nt < 4; ++nt) {
      v8f acc = {};
#pragma unroll
      for (int kt = 0; kt < 2; ++kt) acc = wmma_bf16(a[kt], load_b_frag(n2T, 72, nt * 16, kt * 32), acc);
      int col = nt * 16 + laneRow;
      float b = nb2s[col];
#pragma unroll
      for (int i = 0; i < 8; ++i) {
        int row = i + laneHi * 8;
        o[row * 68 + col] = gelu_f(acc[i] + b);
      }
    }
  }
  __builtin_amdgcn_wave_barrier();

  if (lane < 16) {
    float mu = 0.f;
    for (int k = 0; k < 64; ++k) mu += o[lane * 68 + k];
    mu *= (1.f / 64.f);
    float var = 0.f;
    for (int k = 0; k < 64; ++k) { float d = o[lane * 68 + k] - mu; var += d * d; }
    var *= (1.f / 64.f);
    mus[wave][lane] = mu;
    rss[wave][lane] = rsqrtf(var + 1e-5f);
  }
  __builtin_amdgcn_wave_barrier();

  for (int j = 0; j < 16; ++j) {
    float mu = mus[wave][j], rs = rss[wave][j];
    int node = rBase + j;
    h_out[node * 64 + lane]      = (o[j * 68 + lane] - mu) * rs * lngs[lane] + lnbs[lane];
    h_out[node * 64 + 32 + lane] = (o[j * 68 + 32 + lane] - mu) * rs * lngs[32 + lane] + lnbs[32 + lane];
  }
}

// ---------------- output head: [N,64] @ [64,32] + b ----------------
__global__ __launch_bounds__(128) void egnn_out_kernel(
    const float* __restrict__ h, const float* __restrict__ ow,
    const float* __restrict__ ob, float* __restrict__ out) {
  __shared__ __align__(16) __bf16 oT[32 * 72];
  __shared__ float obs[32];
  __shared__ __align__(16) __bf16 aT[OK_WAVES][16 * 72];

  const int tid = threadIdx.x;
  for (int i = tid; i < 64 * 32; i += 128) {
    int k = i >> 5, n = i & 31;
    oT[n * 72 + k] = f2bf(ow[i]);
  }
  if (tid < 32) obs[tid] = ob[tid];
  __syncthreads();

  const int wave = tid >> 5, lane = tid & 31;
  const int laneRow = lane & 15, laneHi = lane >> 4;
  const int tile = blockIdx.x * OK_WAVES + wave;
  if (tile >= NODE_TILES) return;
  const int rBase = tile * 16;
  __bf16* a_tile = aT[wave];

  for (int j = 0; j < 16; ++j) {
    int node = rBase + j;
    a_tile[j * 72 + lane]      = f2bf(h[node * 64 + lane]);
    a_tile[j * 72 + 32 + lane] = f2bf(h[node * 64 + 32 + lane]);
  }
  __builtin_amdgcn_wave_barrier();

  v16bf a[2];
#pragma unroll
  for (int kt = 0; kt < 2; ++kt) a[kt] = load_a_frag(a_tile, 72, kt * 32);
#pragma unroll
  for (int nt = 0; nt < 2; ++nt) {
    v8f acc = {};
#pragma unroll
    for (int kt = 0; kt < 2; ++kt) acc = wmma_bf16(a[kt], load_b_frag(oT, 72, nt * 16, kt * 32), acc);
    int col = nt * 16 + laneRow;
    float b = obs[col];
#pragma unroll
    for (int i = 0; i < 8; ++i) {
      int row = i + laneHi * 8;
      out[(rBase + row) * 32 + col] = acc[i] + b;
    }
  }
}

extern "C" void kernel_launch(void* const* d_in, const int* in_sizes, int n_in,
                              void* d_out, int out_size, void* d_ws, size_t ws_size,
                              hipStream_t stream) {
  (void)in_sizes; (void)n_in; (void)out_size; (void)ws_size;
  const float* node_feat = (const float*)d_in[0];
  const float* xyz       = (const float*)d_in[1];
  const int*   src       = (const int*)d_in[2];
  const int*   dst       = (const int*)d_in[3];
  const float* edge_w1   = (const float*)d_in[4];
  const float* edge_b1   = (const float*)d_in[5];
  const float* edge_w2   = (const float*)d_in[6];
  const float* edge_b2   = (const float*)d_in[7];
  const float* coord_w1  = (const float*)d_in[8];
  const float* coord_b1  = (const float*)d_in[9];
  const float* coord_w2  = (const float*)d_in[10];
  const float* node_w1   = (const float*)d_in[11];
  const float* node_b1   = (const float*)d_in[12];
  const float* node_w2   = (const float*)d_in[13];
  const float* node_b2   = (const float*)d_in[14];
  const float* ln_g      = (const float*)d_in[15];
  const float* ln_b      = (const float*)d_in[16];
  const float* out_w     = (const float*)d_in[17];
  const float* out_b     = (const float*)d_in[18];

  float* ws = (float*)d_ws;
  const size_t NN = (size_t)N_NODES;
  float* h1 = ws;
  float* h2 = h1 + NN * 64;
  float* x1 = h2 + NN * 64;
  float* x2 = x1 + NN * 3;
  float* hn = x2 + NN * 3;
  float* xs = hn + NN * 64;
  float* dg = xs + NN * 3;   // hn..dg contiguous: N*68 floats

  const int edgeBlocks = N_EDGES / (16 * EK_WAVES);                 // 25000
  const int nodeBlocks = (NODE_TILES + NK_WAVES - 1) / NK_WAVES;    // 1563
  const int outBlocks  = (NODE_TILES + OK_WAVES - 1) / OK_WAVES;    // 1563
  const int zeroCount  = N_NODES * 68;
  const int zeroBlocks = (zeroCount + 255) / 256;

  const float* hcur = node_feat;
  const float* xcur = xyz;
  float* houts[2] = { h1, h2 };
  float* xouts[2] = { x1, x2 };

  for (int l = 0; l < 2; ++l) {
    zero_kernel<<<zeroBlocks, 256, 0, stream>>>(hn, zeroCount);
    egnn_edge_kernel<<<edgeBlocks, 128, 0, stream>>>(
        hcur, xcur, src, dst,
        edge_w1 + (size_t)l * 129 * 64, edge_b1 + (size_t)l * 64,
        edge_w2 + (size_t)l * 64 * 64,  edge_b2 + (size_t)l * 64,
        coord_w1 + (size_t)l * 64 * 64, coord_b1 + (size_t)l * 64,
        coord_w2 + (size_t)l * 64,
        hn, xs, dg);
    egnn_node_kernel<<<nodeBlocks, 128, 0, stream>>>(
        hcur, xcur, hn, xs, dg,
        node_w1 + (size_t)l * 128 * 64, node_b1 + (size_t)l * 64,
        node_w2 + (size_t)l * 64 * 64,  node_b2 + (size_t)l * 64,
        ln_g, ln_b, houts[l], xouts[l]);
    hcur = houts[l];
    xcur = xouts[l];
  }
  egnn_out_kernel<<<outBlocks, 128, 0, stream>>>(hcur, out_w, out_b, (float*)d_out);
}
